// STUFilterApprox_41601053229842
// MI455X (gfx1250) — matrix operand order repaired and checked by reading, not compiled, not run on hardware
//
#include <hip/hip_runtime.h>
#include <hip/hip_bf16.h>

typedef __attribute__((ext_vector_type(16))) __bf16 v16bf;
typedef __attribute__((ext_vector_type(8)))  __bf16 v8bf;
typedef __attribute__((ext_vector_type(4)))  __bf16 v4bf;
typedef __attribute__((ext_vector_type(8)))  float  v8f;

#define Bn 8
#define Tn 4096
#define Nn 1024
#define Cn 512
#define Rr 24
#define NF 8192

#define PI2 6.28318530717958647692f

__device__ __forceinline__ v16bf cat8(v8bf a, v8bf b) {
  return __builtin_shufflevector(a, b, 0, 1, 2, 3, 4, 5, 6, 7,
                                 8, 9, 10, 11, 12, 13, 14, 15);
}

// ---------------------------------------------------------------------------
// One-time fp32 -> bf16 hi/lo split (x ~= hi + lo). Removes all conversion
// VALU from the GEMM inner loops; same total bytes as the fp32 original.
// ---------------------------------------------------------------------------
__global__ __launch_bounds__(256) void split_kernel(
    const float* __restrict__ x, __bf16* __restrict__ hi,
    __bf16* __restrict__ lo, int n4) {
  int i = blockIdx.x * 256 + threadIdx.x;
  if (i >= n4) return;
  float4 v = ((const float4*)x)[i];
  float a[4] = {v.x, v.y, v.z, v.w};
  v4bf h4, l4;
#pragma unroll
  for (int e = 0; e < 4; ++e) {
    __bf16 hh = (__bf16)a[e];
    h4[e] = hh;
    l4[e] = (__bf16)(a[e] - (float)hh);
  }
  ((v4bf*)hi)[i] = h4;
  ((v4bf*)lo)[i] = l4;
}

// ---------------------------------------------------------------------------
// GEMM 1: z_t[b,c,t] = sum_n W_branch[c,n] * h[b,t,n]
// Wave computes a 16x64 strip: M=c, N=t (4 subtiles), K=n.
// All operands pre-split bf16: inner loop = b128 loads + WMMA only.
// ---------------------------------------------------------------------------
__global__ __launch_bounds__(256) void gemm_branch_kernel(
    const __bf16* __restrict__ hHi, const __bf16* __restrict__ hLo,
    const __bf16* __restrict__ WbHi, const __bf16* __restrict__ WbLo,
    float* __restrict__ z) {
  int wid  = (blockIdx.x * 256 + threadIdx.x) >> 5;   // 0..16383
  int lane = threadIdx.x & 31;
  int tq = wid & 63;           // t quad-tile (T/64)
  int ct = (wid >> 6) & 31;    // c tile (C/16)
  int b  = wid >> 11;          // batch
  int c0 = ct << 4, t0 = tq << 6;

  int l15 = lane & 15;
  bool hiHalf = lane >= 16;
  int akg = hiHalf ? 8 : 0;    // 16-bit A layout: lanes16-31 start at K=8
  int bkg = hiHalf ? 16 : 0;   // unused: B is loaded with akg, same K order as A

  const __bf16* Ah = WbHi + (size_t)(c0 + l15) * Nn;
  const __bf16* Al = WbLo + (size_t)(c0 + l15) * Nn;
  const __bf16* Bh0 = hHi + ((size_t)(b * Tn + t0 + l15)) * Nn;
  const __bf16* Bl0 = hLo + ((size_t)(b * Tn + t0 + l15)) * Nn;

  v8f acc[4] = {};
  for (int n0 = 0; n0 < Nn; n0 += 32) {
    // A tile: two aligned 8-element runs per half (K=akg+e, 16+akg+e)
    v16bf ahi = cat8(*(const v8bf*)(Ah + n0 + akg),
                     *(const v8bf*)(Ah + n0 + 16 + akg));
    v16bf alo = cat8(*(const v8bf*)(Al + n0 + akg),
                     *(const v8bf*)(Al + n0 + 16 + akg));
#pragma unroll
    for (int s = 0; s < 4; ++s) {
      const __bf16* Bh = Bh0 + (size_t)s * 16 * Nn;
      const __bf16* Bl = Bl0 + (size_t)s * 16 * Nn;
      v16bf bhi = cat8(*(const v8bf*)(Bh + n0 + akg),
                       *(const v8bf*)(Bh + n0 + 16 + akg));
      v16bf blo = cat8(*(const v8bf*)(Bl + n0 + akg),
                       *(const v8bf*)(Bl + n0 + 16 + akg));
      acc[s] = __builtin_amdgcn_wmma_f32_16x16x32_bf16(false, ahi, false, bhi, (short)0, acc[s], false, false);
      acc[s] = __builtin_amdgcn_wmma_f32_16x16x32_bf16(false, ahi, false, blo, (short)0, acc[s], false, false);
      acc[s] = __builtin_amdgcn_wmma_f32_16x16x32_bf16(false, alo, false, bhi, (short)0, acc[s], false, false);
    }
  }

  int mofs = hiHalf ? 8 : 0;   // C/D layout: VGPR i holds M=i / M=8+i
#pragma unroll
  for (int s = 0; s < 4; ++s)
#pragma unroll
    for (int i = 0; i < 8; ++i)
      z[((size_t)(b * Cn + c0 + mofs + i)) * Tn + t0 + s * 16 + l15] = acc[s][i];
}

// ---------------------------------------------------------------------------
// Mixed-radix FFT in LDS: radix-2 stage (len=8192) + 6 radix-4 stages.
// DIF: natural -> digit-reversed. Inverse = conjugate-transposed stages in
// reverse order. Pointwise multiply in between is order-independent.
// ---------------------------------------------------------------------------
__device__ __forceinline__ void fft_dif(float2* s) {
  { // radix-2, len = NF, twiddle after butterfly
    const int half = NF / 2;
    const float w0 = -PI2 / (float)NF;
    for (int j = threadIdx.x; j < half; j += 256) {
      float2 a = s[j], b = s[j + half];
      s[j] = make_float2(a.x + b.x, a.y + b.y);
      float tr = a.x - b.x, ti = a.y - b.y;
      float sw, cw; __sincosf(w0 * (float)j, &sw, &cw);
      s[j + half] = make_float2(tr * cw - ti * sw, tr * sw + ti * cw);
    }
    __syncthreads();
  }
  for (int len = NF / 2; len >= 4; len >>= 2) {   // 4096,1024,256,64,16,4
    int q = len >> 2;
    float w0 = -PI2 / (float)len;
    for (int u = threadIdx.x; u < NF / 4; u += 256) {
      int j  = u & (q - 1);
      int i0 = (u << 2) - 3 * j;     // group*len + j
      int i1 = i0 + q, i2 = i0 + 2 * q, i3 = i0 + 3 * q;
      float2 a = s[i0], b = s[i1], c = s[i2], d = s[i3];
      float t0r = a.x + c.x, t0i = a.y + c.y;
      float t1r = a.x - c.x, t1i = a.y - c.y;
      float t2r = b.x + d.x, t2i = b.y + d.y;
      float t3r = b.x - d.x, t3i = b.y - d.y;
      float sw, cw; __sincosf(w0 * (float)j, &sw, &cw);
      float c2 = cw * cw - sw * sw, s2 = 2.f * cw * sw;     // w^2
      float c3 = c2 * cw - s2 * sw, s3 = c2 * sw + s2 * cw; // w^3
      s[i0] = make_float2(t0r + t2r, t0i + t2i);
      float o1r = t1r + t3i, o1i = t1i - t3r;               // (t1 - i t3)
      s[i1] = make_float2(o1r * cw - o1i * sw, o1r * sw + o1i * cw);
      float o2r = t0r - t2r, o2i = t0i - t2i;
      s[i2] = make_float2(o2r * c2 - o2i * s2, o2r * s2 + o2i * c2);
      float o3r = t1r - t3i, o3i = t1i + t3r;               // (t1 + i t3)
      s[i3] = make_float2(o3r * c3 - o3i * s3, o3r * s3 + o3i * c3);
    }
    __syncthreads();
  }
}

__device__ __forceinline__ void fft_dit_inv(float2* s) {
  for (int len = 4; len <= NF / 2; len <<= 2) {   // 4,16,64,256,1024,4096
    int q = len >> 2;
    float w0 = PI2 / (float)len;                  // conjugate twiddles
    for (int u = threadIdx.x; u < NF / 4; u += 256) {
      int j  = u & (q - 1);
      int i0 = (u << 2) - 3 * j;
      int i1 = i0 + q, i2 = i0 + 2 * q, i3 = i0 + 3 * q;
      float2 y0 = s[i0], y1 = s[i1], y2 = s[i2], y3 = s[i3];
      float sw, cw; __sincosf(w0 * (float)j, &sw, &cw);
      float c2 = cw * cw - sw * sw, s2 = 2.f * cw * sw;
      float c3 = c2 * cw - s2 * sw, s3 = c2 * sw + s2 * cw;
      float u1r = y1.x * cw - y1.y * sw, u1i = y1.x * sw + y1.y * cw;
      float u2r = y2.x * c2 - y2.y * s2, u2i = y2.x * s2 + y2.y * c2;
      float u3r = y3.x * c3 - y3.y * s3, u3i = y3.x * s3 + y3.y * c3;
      float t0r = y0.x + u2r, t0i = y0.y + u2i;
      float t1r = y0.x - u2r, t1i = y0.y - u2i;
      float t2r = u1r + u3r,  t2i = u1i + u3i;
      float t3r = u1r - u3r,  t3i = u1i - u3i;
      s[i0] = make_float2(t0r + t2r, t0i + t2i);
      s[i1] = make_float2(t1r - t3i, t1i + t3r);            // t1 + i t3
      s[i2] = make_float2(t0r - t2r, t0i - t2i);
      s[i3] = make_float2(t1r + t3i, t1i - t3r);            // t1 - i t3
    }
    __syncthreads();
  }
  { // radix-2, len = NF, conjugate twiddle before butterfly
    const int half = NF / 2;
    const float w0 = PI2 / (float)NF;
    for (int j = threadIdx.x; j < half; j += 256) {
      float sw, cw; __sincosf(w0 * (float)j, &sw, &cw);
      float2 a = s[j], b = s[j + half];
      float tr = b.x * cw - b.y * sw, ti = b.x * sw + b.y * cw;
      s[j]        = make_float2(a.x + tr, a.y + ti);
      s[j + half] = make_float2(a.x - tr, a.y - ti);
    }
    __syncthreads();
  }
}

// ---------------------------------------------------------------------------
// Filter spectra (fused tiny GEMM + DIF FFT), stored digit-reversed.
// ---------------------------------------------------------------------------
__global__ __launch_bounds__(256) void filter_fft_kernel(
    const float* __restrict__ phi, const float* __restrict__ fp,
    float2* __restrict__ Ff) {
  __shared__ float2 s[NF];
  int c = blockIdx.x;
  for (int t = threadIdx.x; t < Tn; t += 256) {
    float acc = 0.f;
#pragma unroll
    for (int j = 0; j < Rr; ++j)
      acc += phi[t * Rr + j] * fp[j * Cn + c];
    s[t] = make_float2(acc, 0.f);
  }
  for (int t = Tn + threadIdx.x; t < NF; t += 256)
    s[t] = make_float2(0.f, 0.f);
  __syncthreads();
  fft_dif(s);
  float2* dst = Ff + (size_t)c * NF;
  for (int i = threadIdx.x; i < NF; i += 256) dst[i] = s[i];
}

// ---------------------------------------------------------------------------
// Fused causal conv, batch-pair packed (same real filter => pack two batches
// as re/im of one complex FFT). In-place over z.
// ---------------------------------------------------------------------------
__global__ __launch_bounds__(256) void conv_kernel(
    float* __restrict__ zy, const float2* __restrict__ Ff) {
  __shared__ float2 s[NF];
  int p = blockIdx.x >> 9;     // batch pair (B/2 = 4)
  int c = blockIdx.x & 511;
  float* row0 = zy + ((size_t)((2 * p)     * Cn + c)) * Tn;
  float* row1 = zy + ((size_t)((2 * p + 1) * Cn + c)) * Tn;
  for (int i = threadIdx.x; i < Tn; i += 256)
    s[i] = make_float2(row0[i], row1[i]);
  for (int i = Tn + threadIdx.x; i < NF; i += 256)
    s[i] = make_float2(0.f, 0.f);
  __syncthreads();

  fft_dif(s);

  const float2* F = Ff + (size_t)c * NF;
  for (int i = threadIdx.x; i < NF; i += 256) {
    float2 f = F[i];
    float2 zf = s[i];
    s[i] = make_float2(zf.x * f.x - zf.y * f.y, zf.x * f.y + zf.y * f.x);
  }
  __syncthreads();

  fft_dit_inv(s);

  const float scale = 1.0f / (float)NF;
  for (int i = threadIdx.x; i < Tn; i += 256) {
    row0[i] = s[i].x * scale;
    row1[i] = s[i].y * scale;
  }
}

// ---------------------------------------------------------------------------
// GEMM 2: out[b,t,n] = sum_c y_t[b,c,t] * W_out[n,c]
// Wave computes a 16x64 strip: M=t, N=n (4 subtiles), K=c.
// A (y, fp32) converted once per K-step and reused; B pre-split bf16.
// ---------------------------------------------------------------------------
__global__ __launch_bounds__(256) void gemm_out_kernel(
    const float* __restrict__ y,
    const __bf16* __restrict__ WoHi, const __bf16* __restrict__ WoLo,
    float* __restrict__ out) {
  int wid  = (blockIdx.x * 256 + threadIdx.x) >> 5;   // 0..32767
  int lane = threadIdx.x & 31;
  int nq = wid & 15;           // n quad-tile (N/64)
  int tt = (wid >> 4) & 255;   // t tile
  int b  = wid >> 12;          // batch
  int t0 = tt << 4, n0 = nq << 6;

  int l15 = lane & 15;
  bool hiHalf = lane >= 16;
  int akg = hiHalf ? 8 : 0;
  int bkg = hiHalf ? 16 : 0;

  const float* Abase = y + (size_t)b * Cn * Tn + t0 + l15;  // + K*Tn

  v8f acc[4] = {};
  for (int c0 = 0; c0 < Cn; c0 += 32) {
    // ---- A tile (strided fp32): convert once, reuse for 4 subtiles ----
    float af[16];
#pragma unroll
    for (int e = 0; e < 16; ++e) {
      int K = ((e >> 3) << 4) + akg + (e & 7);
      af[e] = Abase[(size_t)(c0 + K) * Tn];
    }
    v16bf ahi, alo;
#pragma unroll
    for (int e = 0; e < 16; ++e) {
      __bf16 h1 = (__bf16)af[e];
      ahi[e] = h1;
      alo[e] = (__bf16)(af[e] - (float)h1);
    }
#pragma unroll
    for (int s = 0; s < 4; ++s) {
      const __bf16* Bh = WoHi + (size_t)(n0 + s * 16 + l15) * Cn;
      const __bf16* Bl = WoLo + (size_t)(n0 + s * 16 + l15) * Cn;
      v16bf bhi = cat8(*(const v8bf*)(Bh + c0 + akg),
                       *(const v8bf*)(Bh + c0 + 16 + akg));
      v16bf blo = cat8(*(const v8bf*)(Bl + c0 + akg),
                       *(const v8bf*)(Bl + c0 + 16 + akg));
      acc[s] = __builtin_amdgcn_wmma_f32_16x16x32_bf16(false, ahi, false, bhi, (short)0, acc[s], false, false);
      acc[s] = __builtin_amdgcn_wmma_f32_16x16x32_bf16(false, ahi, false, blo, (short)0, acc[s], false, false);
      acc[s] = __builtin_amdgcn_wmma_f32_16x16x32_bf16(false, alo, false, bhi, (short)0, acc[s], false, false);
    }
  }

  int mofs = hiHalf ? 8 : 0;
#pragma unroll
  for (int s = 0; s < 4; ++s)
#pragma unroll
    for (int i = 0; i < 8; ++i)
      out[(size_t)(b * Tn + t0 + mofs + i) * Nn + n0 + s * 16 + l15] = acc[s][i];
}

// ---------------------------------------------------------------------------
extern "C" void kernel_launch(void* const* d_in, const int* in_sizes, int n_in,
                              void* d_out, int out_size, void* d_ws, size_t ws_size,
                              hipStream_t stream) {
  (void)in_sizes; (void)n_in; (void)out_size; (void)ws_size;
  const float* h   = (const float*)d_in[0];   // (B,T,N)
  const float* phi = (const float*)d_in[1];   // (T,r)
  const float* Wb  = (const float*)d_in[2];   // (C,N)
  const float* fp  = (const float*)d_in[3];   // (r,C)
  const float* Wo  = (const float*)d_in[4];   // (N,C)
  float* out = (float*)d_out;                 // (B,T,N)

  char* w = (char*)d_ws;
  float*  z_ws = (float*)w;  w += (size_t)Bn * Cn * Tn * 4;   // 67 MB (z, then y)
  float2* Ff   = (float2*)w; w += (size_t)Cn * NF * 8;        // 33.5 MB
  __bf16* hHi  = (__bf16*)w; w += (size_t)Bn * Tn * Nn * 2;   // 67 MB
  __bf16* hLo  = (__bf16*)w; w += (size_t)Bn * Tn * Nn * 2;   // 67 MB
  __bf16* WbHi = (__bf16*)w; w += (size_t)Cn * Nn * 2;        // 1 MB
  __bf16* WbLo = (__bf16*)w; w += (size_t)Cn * Nn * 2;
  __bf16* WoHi = (__bf16*)w; w += (size_t)Nn * Cn * 2;
  __bf16* WoLo = (__bf16*)w; w += (size_t)Nn * Cn * 2;

  // 0) one-time hi/lo bf16 splits
  split_kernel<<<(Bn * Tn * Nn / 4 + 255) / 256, 256, 0, stream>>>(h, hHi, hLo, Bn * Tn * Nn / 4);
  split_kernel<<<(Cn * Nn / 4 + 255) / 256, 256, 0, stream>>>(Wb, WbHi, WbLo, Cn * Nn / 4);
  split_kernel<<<(Nn * Cn / 4 + 255) / 256, 256, 0, stream>>>(Wo, WoHi, WoLo, Nn * Cn / 4);

  // 1) z_t = W_branch x h^T : 16384 waves (16x64 strips), 8 waves/block
  gemm_branch_kernel<<<2048, 256, 0, stream>>>(hHi, hLo, WbHi, WbLo, z_ws);
  // 2) filter spectra (fused tiny GEMM + DIF FFT), digit-reversed domain
  filter_fft_kernel<<<Cn, 256, 0, stream>>>(phi, fp, Ff);
  // 3) fused FFT -> multiply -> iFFT, batch-pair packed, in-place
  conv_kernel<<<(Bn / 2) * Cn, 256, 0, stream>>>(z_ws, Ff);
  // 4) out = y x W_out^T : 32768 waves (16x64 strips)
  gemm_out_kernel<<<4096, 256, 0, stream>>>(z_ws, WoHi, WoLo, out);
}
